// QGNN_Rot_PermInv_Layer_28217935135258
// MI455X (gfx1250) — matrix operand order repaired and checked by reading, not compiled, not run on hardware
//
#include <hip/hip_runtime.h>
#include <hip/hip_bf16.h>
#include <math.h>

typedef _Float16 v16h __attribute__((ext_vector_type(16)));
typedef float    v8f  __attribute__((ext_vector_type(8)));

#define HID   128
#define INPAD 64

union Frag16 {
    v16h  v;
    float4 f4[2];
};

// -------------------------------------------------------------------------
// Convert fp32 weights to transposed, K-padded fp16 layouts for WMMA B-frags.
// W1t[o][k] : 128 x 64 (k<38 valid), W2t[o][k] : 128 x 128, W3t[o][k] : 16 x 128
// -------------------------------------------------------------------------
__global__ void convert_weights(const float* __restrict__ W1,
                                const float* __restrict__ W2,
                                const float* __restrict__ W3,
                                _Float16* __restrict__ W1t,
                                _Float16* __restrict__ W2t,
                                _Float16* __restrict__ W3t)
{
    int i = blockIdx.x * blockDim.x + threadIdx.x;
    if (i < 128 * INPAD) {
        int o = i / INPAD, k = i % INPAD;
        W1t[o * INPAD + k] = (_Float16)((k < 38) ? W1[k * HID + o] : 0.0f);
        return;
    }
    i -= 128 * INPAD;
    if (i < HID * HID) {
        int o = i / HID, k = i % HID;
        W2t[o * HID + k] = (_Float16)W2[k * HID + o];
        return;
    }
    i -= HID * HID;
    if (i < 16 * HID) {
        int o = i / HID, k = i % HID;
        W3t[o * HID + k] = (_Float16)W3[k * 16 + o];
    }
}

__global__ void zero_ws(float* __restrict__ agg, int* __restrict__ counts, int N)
{
    int i = blockIdx.x * blockDim.x + threadIdx.x;
    if (i < N * 16) agg[i] = 0.0f;
    if (i < N)      counts[i] = 0;
}

__global__ void count_recv(const int* __restrict__ ei, int* __restrict__ counts, int E)
{
    int i = blockIdx.x * blockDim.x + threadIdx.x;
    if (i < E) atomicAdd(&counts[ei[E + i]], 1);
}

// -------------------------------------------------------------------------
// Per-edge MLP: 4 waves/block, 32 edges/wave = 2 M-tiles per wave so each
// B fragment load feeds two v_wmma_f32_16x16x32_f16 issues.
// -------------------------------------------------------------------------
__launch_bounds__(128)
__global__ void edge_mlp(const float* __restrict__ h,
                         const float* __restrict__ npar,
                         const float* __restrict__ epar,
                         const int*   __restrict__ ei,       // flat [2*E]
                         const _Float16* __restrict__ W1t,
                         const _Float16* __restrict__ W2t,
                         const _Float16* __restrict__ W3t,
                         const float* __restrict__ b1,
                         const float* __restrict__ b2,
                         const float* __restrict__ b3,
                         float* __restrict__ agg,
                         int E)
{
    __shared__ __align__(16) _Float16 bufX[4][32][INPAD];  // 16 KB
    __shared__ __align__(16) _Float16 bufH[4][32][HID];    // 32 KB

    const int lane = threadIdx.x & 31;
    const int wv   = threadIdx.x >> 5;                      // 0..3
    const int edgeBase = (blockIdx.x * 4 + wv) * 32;

    // ---- gather 38-dim edge features into LDS (zero-padded to 64) ----
    for (int i = lane; i < 32 * INPAD; i += 32) {
        int e  = i >> 6;
        int f  = i & 63;
        int eg = edgeBase + e;
        float val = 0.0f;
        if (eg < E && f < 38) {
            int s = ei[eg];
            int r = ei[E + eg];
            if      (f < 16) val = h[s * 16 + f];
            else if (f < 32) val = h[r * 16 + (f - 16)];
            else if (f < 34) val = npar[s * 2 + (f - 32)];
            else if (f < 36) val = npar[r * 2 + (f - 34)];
            else if (f == 36) val = epar[ei[2 * eg]];       // torch view(E,-1) quirk
            else              val = epar[ei[2 * eg + 1]];   // collapses to flat index
        }
        bufX[wv][e][f] = (_Float16)val;
    }
    __syncthreads();

    const int m  = lane & 15;   // A-row (within tile) / B-col / C-col for this lane
    const int kh = lane >> 4;   // K-half selector

    // ---- layer 1: X[32x64] @ W1[64x128], SiLU -> bufH ----
    #pragma unroll
    for (int nt = 0; nt < 8; ++nt) {
        v8f acc0 = {}, acc1 = {};
        #pragma unroll
        for (int ks = 0; ks < INPAD; ks += 32) {
            Frag16 B;
            const _Float16* bp = W1t + (nt * 16 + m) * INPAD + ks + kh * 8;
            B.f4[0] = *(const float4*)(bp);
            B.f4[1] = *(const float4*)(bp + 16);
            Frag16 A0, A1;
            A0.f4[0] = *(const float4*)&bufX[wv][m][ks + kh * 8];
            A0.f4[1] = *(const float4*)&bufX[wv][m][ks + 16 + kh * 8];
            A1.f4[0] = *(const float4*)&bufX[wv][16 + m][ks + kh * 8];
            A1.f4[1] = *(const float4*)&bufX[wv][16 + m][ks + 16 + kh * 8];
            acc0 = __builtin_amdgcn_wmma_f32_16x16x32_f16(false, A0.v, false, B.v,
                                                          (short)0, acc0, false, false);
            acc1 = __builtin_amdgcn_wmma_f32_16x16x32_f16(false, A1.v, false, B.v,
                                                          (short)0, acc1, false, false);
        }
        float bias = b1[nt * 16 + m];
        #pragma unroll
        for (int r = 0; r < 8; ++r) {
            float x0 = acc0[r] + bias;
            float x1 = acc1[r] + bias;
            bufH[wv][r + 8 * kh][nt * 16 + m]      = (_Float16)(x0 / (1.0f + __expf(-x0)));
            bufH[wv][16 + r + 8 * kh][nt * 16 + m] = (_Float16)(x1 / (1.0f + __expf(-x1)));
        }
    }
    __syncthreads();

    // ---- layer 2: H1[32x128] @ W2[128x128] (all tiles in regs, then writeback) ----
    v8f acc2[8][2];
    #pragma unroll
    for (int nt = 0; nt < 8; ++nt) {
        v8f a0 = {}, a1 = {};
        #pragma unroll
        for (int ks = 0; ks < HID; ks += 32) {
            Frag16 B;
            const _Float16* bp = W2t + (nt * 16 + m) * HID + ks + kh * 8;
            B.f4[0] = *(const float4*)(bp);
            B.f4[1] = *(const float4*)(bp + 16);
            Frag16 A0, A1;
            A0.f4[0] = *(const float4*)&bufH[wv][m][ks + kh * 8];
            A0.f4[1] = *(const float4*)&bufH[wv][m][ks + 16 + kh * 8];
            A1.f4[0] = *(const float4*)&bufH[wv][16 + m][ks + kh * 8];
            A1.f4[1] = *(const float4*)&bufH[wv][16 + m][ks + 16 + kh * 8];
            a0 = __builtin_amdgcn_wmma_f32_16x16x32_f16(false, A0.v, false, B.v,
                                                        (short)0, a0, false, false);
            a1 = __builtin_amdgcn_wmma_f32_16x16x32_f16(false, A1.v, false, B.v,
                                                        (short)0, a1, false, false);
        }
        acc2[nt][0] = a0;
        acc2[nt][1] = a1;
    }
    __syncthreads();
    #pragma unroll
    for (int nt = 0; nt < 8; ++nt) {
        float bias = b2[nt * 16 + m];
        #pragma unroll
        for (int r = 0; r < 8; ++r) {
            float x0 = acc2[nt][0][r] + bias;
            float x1 = acc2[nt][1][r] + bias;
            bufH[wv][r + 8 * kh][nt * 16 + m]      = (_Float16)(x0 / (1.0f + __expf(-x0)));
            bufH[wv][16 + r + 8 * kh][nt * 16 + m] = (_Float16)(x1 / (1.0f + __expf(-x1)));
        }
    }
    __syncthreads();

    // ---- layer 3: H2[32x128] @ W3[128x16] + scatter-add messages ----
    v8f m0 = {}, m1 = {};
    #pragma unroll
    for (int ks = 0; ks < HID; ks += 32) {
        Frag16 B;
        const _Float16* bp = W3t + m * HID + ks + kh * 8;
        B.f4[0] = *(const float4*)(bp);
        B.f4[1] = *(const float4*)(bp + 16);
        Frag16 A0, A1;
        A0.f4[0] = *(const float4*)&bufH[wv][m][ks + kh * 8];
        A0.f4[1] = *(const float4*)&bufH[wv][m][ks + 16 + kh * 8];
        A1.f4[0] = *(const float4*)&bufH[wv][16 + m][ks + kh * 8];
        A1.f4[1] = *(const float4*)&bufH[wv][16 + m][ks + 16 + kh * 8];
        m0 = __builtin_amdgcn_wmma_f32_16x16x32_f16(false, A0.v, false, B.v,
                                                    (short)0, m0, false, false);
        m1 = __builtin_amdgcn_wmma_f32_16x16x32_f16(false, A1.v, false, B.v,
                                                    (short)0, m1, false, false);
    }
    float bias = b3[m];
    #pragma unroll
    for (int r = 0; r < 8; ++r) {
        int e0 = r + 8 * kh;
        int eg0 = edgeBase + e0;
        if (eg0 < E) {
            int rcv = ei[E + eg0];
            atomicAdd(&agg[rcv * 16 + m], m0[r] + bias);
        }
        int eg1 = edgeBase + 16 + e0;
        if (eg1 < E) {
            int rcv = ei[E + eg1];
            atomicAdd(&agg[rcv * 16 + m], m1[r] + bias);
        }
    }
}

// -------------------------------------------------------------------------
// Per-node: mean, zero->I, skew, expm (scaling & squaring + Horner Taylor),
// then U @ h @ U^T.
// -------------------------------------------------------------------------
__device__ __forceinline__ void mm4(const float* A, const float* B, float* C)
{
    #pragma unroll
    for (int r = 0; r < 4; ++r)
        #pragma unroll
        for (int c = 0; c < 4; ++c) {
            float s = 0.0f;
            #pragma unroll
            for (int k = 0; k < 4; ++k) s += A[r * 4 + k] * B[k * 4 + c];
            C[r * 4 + c] = s;
        }
}

__global__ void node_expm(const float* __restrict__ h,
                          const float* __restrict__ agg,
                          const int*   __restrict__ counts,
                          float* __restrict__ out, int N)
{
    int i = blockIdx.x * blockDim.x + threadIdx.x;
    if (i >= N) return;

    int   c   = counts[i];
    float inv = 1.0f / (float)(c > 0 ? c : 1);

    float a[16];
    float asum = 0.0f;
    #pragma unroll
    for (int j = 0; j < 16; ++j) { a[j] = agg[i * 16 + j] * inv; asum += fabsf(a[j]); }
    if (asum == 0.0f) {
        #pragma unroll
        for (int j = 0; j < 16; ++j) a[j] = 0.0f;
        a[0] = a[5] = a[10] = a[15] = 1.0f;
    }

    float S[16];
    #pragma unroll
    for (int r = 0; r < 4; ++r)
        #pragma unroll
        for (int cc = 0; cc < 4; ++cc)
            S[r * 4 + cc] = 0.5f * (a[r * 4 + cc] - a[cc * 4 + r]);

    float nrm = 0.0f;
    #pragma unroll
    for (int j = 0; j < 16; ++j) nrm += fabsf(S[j]);
    int s = 0;
    while (nrm > 0.25f && s < 40) { nrm *= 0.5f; ++s; }
    float sc = 1.0f;
    for (int j = 0; j < s; ++j) sc *= 0.5f;

    float T[16];
    #pragma unroll
    for (int j = 0; j < 16; ++j) T[j] = S[j] * sc;

    // Horner Taylor: R = I + T(I + T/2(I + ... ))
    float R[16] = {1,0,0,0, 0,1,0,0, 0,0,1,0, 0,0,0,1};
    for (int k = 9; k >= 1; --k) {
        float P[16];
        mm4(T, R, P);
        float ik = 1.0f / (float)k;
        #pragma unroll
        for (int j = 0; j < 16; ++j) R[j] = P[j] * ik;
        R[0] += 1.0f; R[5] += 1.0f; R[10] += 1.0f; R[15] += 1.0f;
    }
    for (int j = 0; j < s; ++j) {
        float P[16];
        mm4(R, R, P);
        #pragma unroll
        for (int q = 0; q < 16; ++q) R[q] = P[q];
    }

    float Hm[16];
    #pragma unroll
    for (int j = 0; j < 16; ++j) Hm[j] = h[i * 16 + j];
    float P[16];
    mm4(R, Hm, P);
    // out = P @ R^T
    #pragma unroll
    for (int r = 0; r < 4; ++r)
        #pragma unroll
        for (int cc = 0; cc < 4; ++cc) {
            float sv = 0.0f;
            #pragma unroll
            for (int k = 0; k < 4; ++k) sv += P[r * 4 + k] * R[cc * 4 + k];
            out[i * 16 + r * 4 + cc] = sv;
        }
}

// -------------------------------------------------------------------------
extern "C" void kernel_launch(void* const* d_in, const int* in_sizes, int n_in,
                              void* d_out, int out_size, void* d_ws, size_t ws_size,
                              hipStream_t stream)
{
    const float* h    = (const float*)d_in[0];
    const float* npar = (const float*)d_in[1];
    const float* epar = (const float*)d_in[2];
    const float* W1   = (const float*)d_in[3];
    const float* b1   = (const float*)d_in[4];
    const float* W2   = (const float*)d_in[5];
    const float* b2   = (const float*)d_in[6];
    const float* W3   = (const float*)d_in[7];
    const float* b3   = (const float*)d_in[8];
    const int*   ei   = (const int*)d_in[9];

    const int N = in_sizes[0] / 16;
    const int E = in_sizes[9] / 2;
    float* out = (float*)d_out;

    char* ws = (char*)d_ws;
    _Float16* W1t = (_Float16*)ws;  ws += (size_t)128 * INPAD * 2;
    _Float16* W2t = (_Float16*)ws;  ws += (size_t)HID * HID * 2;
    _Float16* W3t = (_Float16*)ws;  ws += (size_t)16 * HID * 2;
    float* agg    = (float*)ws;     ws += (size_t)N * 16 * 4;
    int*   counts = (int*)ws;

    const int wtot = 128 * INPAD + HID * HID + 16 * HID;
    convert_weights<<<(wtot + 255) / 256, 256, 0, stream>>>(W1, W2, W3, W1t, W2t, W3t);
    zero_ws<<<(N * 16 + 255) / 256, 256, 0, stream>>>(agg, counts, N);
    count_recv<<<(E + 255) / 256, 256, 0, stream>>>(ei, counts, E);

    const int nblk = (E + 127) / 128;   // 4 waves x 32 edges per block
    edge_mlp<<<nblk, 128, 0, stream>>>(h, npar, epar, ei, W1t, W2t, W3t,
                                       b1, b2, b3, agg, E);

    node_expm<<<(N + 255) / 256, 256, 0, stream>>>(h, agg, counts, out, N);
}
